// NodeEmbedderSelfAttention_11304353923850
// MI455X (gfx1250) — matrix operand order repaired and checked by reading, not compiled, not run on hardware
//
#include <hip/hip_runtime.h>

typedef __attribute__((ext_vector_type(16))) _Float16 v16h;
typedef __attribute__((ext_vector_type(8)))  _Float16 v8h;
typedef __attribute__((ext_vector_type(8)))  float    v8f;

#define BATCH 1024
#define SEQL  256
#define DIN   72      // T(8) + feats(64)
#define KPAD  96      // padded K, 3 chunks of 32
#define HDIM  256
#define ODIM  256
#define EDIM  128

__device__ __forceinline__ float rrelu_f(float v) {
    // PyTorch RReLU eval slope = (1/8 + 1/3)/2
    return v >= 0.0f ? v : 0.22916666666666666f * v;
}

// A fragment: 16x32 f16 tile, row m = lane%16, k pattern per ISA:
// VGPR0-3 -> k = kc*32 + 8*half + 0..7 ; VGPR4-7 -> k = kc*32 + 16 + 8*half + 0..7
__device__ __forceinline__ v16h load_a_frag(const _Float16* xrow, int kc, int half) {
    const _Float16* p = xrow + kc * 32 + 8 * half;
    v16h a;
    *(v8h*)&a         = *(const v8h*)(p);
    *(((v8h*)&a) + 1) = *(const v8h*)(p + 16);
    return a;
}

// B fragment: 32x16 f16 tile (KxN), col n = lane%16, k = kc*32 + 8*half + 0..7, then kc*32 + 16 + 8*half + 0..7.
// Gathered from row-major f32 weight [72, 256]; zero-pad k >= 72.
__device__ __forceinline__ v16h load_b_frag(const float* __restrict__ w, int n, int kc, int half) {
    v16h b;
    const int kbase = kc * 32 + 8 * half;
#pragma unroll
    for (int i = 0; i < 16; ++i) {
        const int k = kbase + (i < 8 ? i : 8 + i);
        b[i] = (k < DIN) ? (_Float16)w[k * HDIM + n] : (_Float16)0.0f;
    }
    return b;
}

__device__ __forceinline__ v8f wmma_acc(v16h a, v16h b, v8f c) {
    return __builtin_amdgcn_wmma_f32_16x16x32_f16(false, a, false, b, (short)0, c, false, false);
}

__global__ __launch_bounds__(256)
void attn_kernel(const float* __restrict__ ts, const float* __restrict__ feats,
                 const float* __restrict__ t2v_w, const float* __restrict__ t2v_b,
                 const float* __restrict__ kw, const float* __restrict__ kb,
                 const float* __restrict__ qw, const float* __restrict__ qb,
                 const float* __restrict__ vw, const float* __restrict__ vb,
                 const float* __restrict__ ow, const float* __restrict__ ob,
                 const int*   __restrict__ lens, float* __restrict__ out) {
    __shared__ _Float16 x_lds[SEQL * KPAD];   // 48 KB
    __shared__ float scores_lds[SEQL];
    __shared__ float a_lds[SEQL];
    __shared__ float q_lds[HDIM];
    __shared__ float ctx_lds[HDIM];
    __shared__ float red_lds[SEQL];

    const int b    = blockIdx.x;
    const int tid  = threadIdx.x;
    const int lane = tid & 31;
    const int wave = tid >> 5;
    const int half = lane >> 4;
    const int ln16 = lane & 15;
    const int len  = lens[b];

    // Prefetch weight streams (L2-resident, reused by all 1024 blocks)
    __builtin_prefetch(ow + tid * 64, 0, 1);

    // ---- build x = [time2vec | feats] as f16, zero-padded K -> 96 ----
    {
        const int l = tid;                         // one row per thread
        _Float16* xr = x_lds + l * KPAD;
        const float t = ts[b * SEQL + l];
#pragma unroll
        for (int d = 0; d < 8; ++d) {
            const float z = t * t2v_w[d] + t2v_b[d];
            xr[d] = (_Float16)(d == 0 ? z : __sinf(z));
        }
        const float* fr = feats + ((size_t)b * SEQL + l) * 64;
#pragma unroll 8
        for (int d = 0; d < 64; ++d) xr[8 + d] = (_Float16)fr[d];
#pragma unroll
        for (int d = DIN; d < KPAD; ++d) xr[d] = (_Float16)0.0f;
        scores_lds[tid] = 0.0f;
    }
    __syncthreads();

    const int nt0 = wave * 2;
    const int nt1 = wave * 2 + 1;
    const int n0  = nt0 * 16 + ln16;
    const int n1  = nt1 * 16 + ln16;

    // ================= Q pass: q = colsum(rrelu(x@qw+qb) masked) / (len*sqrt(H)) ===========
    {
        v16h bq0[3], bq1[3];
#pragma unroll
        for (int kc = 0; kc < 3; ++kc) { bq0[kc] = load_b_frag(qw, n0, kc, half);
                                         bq1[kc] = load_b_frag(qw, n1, kc, half); }
        const float bias0 = qb[n0], bias1 = qb[n1];
        float cs0 = 0.0f, cs1 = 0.0f;
        for (int rt = 0; rt < 16; ++rt) {
            const _Float16* xrow = x_lds + (rt * 16 + ln16) * KPAD;
            v16h a0 = load_a_frag(xrow, 0, half);
            v16h a1 = load_a_frag(xrow, 1, half);
            v16h a2 = load_a_frag(xrow, 2, half);
            v8f c0, c1;
#pragma unroll
            for (int i = 0; i < 8; ++i) { c0[i] = bias0; c1[i] = bias1; }
            c0 = wmma_acc(a0, bq0[0], c0); c0 = wmma_acc(a1, bq0[1], c0); c0 = wmma_acc(a2, bq0[2], c0);
            c1 = wmma_acc(a0, bq1[0], c1); c1 = wmma_acc(a1, bq1[1], c1); c1 = wmma_acc(a2, bq1[2], c1);
#pragma unroll
            for (int r = 0; r < 8; ++r) {
                const int m = rt * 16 + r + 8 * half;
                const float vmask = (m < len) ? 1.0f : 0.0f;   // branchless (v_cndmask)
                cs0 += vmask * rrelu_f(c0[r]);
                cs1 += vmask * rrelu_f(c1[r]);
            }
        }
        const float inv = 1.0f / ((float)len * 16.0f);   // 1/(len*sqrt(256))
        const float t0 = cs0 + __shfl_xor(cs0, 16, 32);
        const float t1 = cs1 + __shfl_xor(cs1, 16, 32);
        if (lane < 16) { q_lds[nt0 * 16 + lane] = t0 * inv;
                         q_lds[nt1 * 16 + lane] = t1 * inv; }
    }
    __syncthreads();

    // ================= K pass: scores[m] += sum_n rrelu(K)[m,n] * q[n] =====================
    {
        v16h bk0[3], bk1[3];
#pragma unroll
        for (int kc = 0; kc < 3; ++kc) { bk0[kc] = load_b_frag(kw, n0, kc, half);
                                         bk1[kc] = load_b_frag(kw, n1, kc, half); }
        const float bias0 = kb[n0], bias1 = kb[n1];
        const float qv0 = q_lds[n0], qv1 = q_lds[n1];
        for (int rt = 0; rt < 16; ++rt) {
            const _Float16* xrow = x_lds + (rt * 16 + ln16) * KPAD;
            v16h a0 = load_a_frag(xrow, 0, half);
            v16h a1 = load_a_frag(xrow, 1, half);
            v16h a2 = load_a_frag(xrow, 2, half);
            v8f c0, c1;
#pragma unroll
            for (int i = 0; i < 8; ++i) { c0[i] = bias0; c1[i] = bias1; }
            c0 = wmma_acc(a0, bk0[0], c0); c0 = wmma_acc(a1, bk0[1], c0); c0 = wmma_acc(a2, bk0[2], c0);
            c1 = wmma_acc(a0, bk1[0], c1); c1 = wmma_acc(a1, bk1[1], c1); c1 = wmma_acc(a2, bk1[2], c1);
            float mine = 0.0f;
#pragma unroll
            for (int r = 0; r < 8; ++r) {
                const int m = rt * 16 + r + 8 * half;
                float p = rrelu_f(c0[r]) * qv0 + rrelu_f(c1[r]) * qv1;
                p = (m < len) ? p : 0.0f;                      // branchless
                p += __shfl_xor(p, 1, 32); p += __shfl_xor(p, 2, 32);
                p += __shfl_xor(p, 4, 32); p += __shfl_xor(p, 8, 32);
                mine = (ln16 == r) ? p : mine;                 // branchless select
            }
            if (ln16 < 8) atomicAdd(&scores_lds[rt * 16 + (lane & 7) + 8 * half], mine);
        }
    }
    __syncthreads();

    // ================= masked softmax over L ==============================================
    {
        const int m = tid;
        const bool valid = m < len;
        const float s = valid ? scores_lds[m] : -3.0e38f;
        red_lds[tid] = s; __syncthreads();
        for (int off = 128; off > 0; off >>= 1) {
            if (tid < off) red_lds[tid] = fmaxf(red_lds[tid], red_lds[tid + off]);
            __syncthreads();
        }
        const float mx = red_lds[0]; __syncthreads();
        const float e = valid ? __expf(s - mx) : 0.0f;
        red_lds[tid] = e; __syncthreads();
        for (int off = 128; off > 0; off >>= 1) {
            if (tid < off) red_lds[tid] += red_lds[tid + off];
            __syncthreads();
        }
        const float invs = 1.0f / red_lds[0]; __syncthreads();
        a_lds[tid] = e * invs;     // exactly 0 at pad positions
    }
    __syncthreads();

    // ================= V pass: ctx[n] = sum_m a[m] * rrelu(V)[m,n] =========================
    {
        v16h bv0[3], bv1[3];
#pragma unroll
        for (int kc = 0; kc < 3; ++kc) { bv0[kc] = load_b_frag(vw, n0, kc, half);
                                         bv1[kc] = load_b_frag(vw, n1, kc, half); }
        const float bias0 = vb[n0], bias1 = vb[n1];
        float cx0 = 0.0f, cx1 = 0.0f;
        for (int rt = 0; rt < 16; ++rt) {
            const _Float16* xrow = x_lds + (rt * 16 + ln16) * KPAD;
            v16h a0 = load_a_frag(xrow, 0, half);
            v16h a1 = load_a_frag(xrow, 1, half);
            v16h a2 = load_a_frag(xrow, 2, half);
            v8f c0, c1;
#pragma unroll
            for (int i = 0; i < 8; ++i) { c0[i] = bias0; c1[i] = bias1; }
            c0 = wmma_acc(a0, bv0[0], c0); c0 = wmma_acc(a1, bv0[1], c0); c0 = wmma_acc(a2, bv0[2], c0);
            c1 = wmma_acc(a0, bv1[0], c1); c1 = wmma_acc(a1, bv1[1], c1); c1 = wmma_acc(a2, bv1[2], c1);
#pragma unroll
            for (int r = 0; r < 8; ++r) {
                const int m = rt * 16 + r + 8 * half;
                const float wgt = a_lds[m];    // already 0 at pad -> masks V implicitly
                cx0 += wgt * rrelu_f(c0[r]);
                cx1 += wgt * rrelu_f(c1[r]);
            }
        }
        const float t0 = cx0 + __shfl_xor(cx0, 16, 32);
        const float t1 = cx1 + __shfl_xor(cx1, 16, 32);
        if (lane < 16) { ctx_lds[nt0 * 16 + lane] = t0;
                         ctx_lds[nt1 * 16 + lane] = t1; }
    }
    __syncthreads();

    // ================= out projection: rrelu(ctx @ ow + ob) ================================
    {
        const int o = tid;
        float acc = ob[o];
#pragma unroll 8
        for (int h = 0; h < HDIM; ++h) acc += ctx_lds[h] * ow[h * ODIM + o];
        out[(size_t)b * ODIM + o] = rrelu_f(acc);
    }
}

__global__ __launch_bounds__(128)
void combo_kernel(const float* __restrict__ fo, const float* __restrict__ ro,
                  const float* __restrict__ cw, const float* __restrict__ cb,
                  float* __restrict__ out) {
    const int b = blockIdx.x;
    const int o = threadIdx.x;     // 128
    __builtin_prefetch(cw + o * 128, 0, 1);
    float acc = cb[o];
    const float* f = fo + (size_t)b * ODIM;
    const float* r = ro + (size_t)b * ODIM;
#pragma unroll 4
    for (int h = 0; h < ODIM; ++h) acc += f[h] * cw[h * EDIM + o];
#pragma unroll 4
    for (int h = 0; h < ODIM; ++h) acc += r[h] * cw[(ODIM + h) * EDIM + o];
    out[(size_t)b * EDIM + o] = 1.0f / (1.0f + __expf(-acc));
}

extern "C" void kernel_launch(void* const* d_in, const int* in_sizes, int n_in,
                              void* d_out, int out_size, void* d_ws, size_t ws_size,
                              hipStream_t stream) {
    (void)in_sizes; (void)n_in; (void)out_size; (void)ws_size;
    const float* ts_f    = (const float*)d_in[0];
    const float* feats_f = (const float*)d_in[1];
    const float* ts_r    = (const float*)d_in[2];
    const float* feats_r = (const float*)d_in[3];
    const float* t2v_f_w = (const float*)d_in[4];
    const float* t2v_f_b = (const float*)d_in[5];
    const float* t2v_r_w = (const float*)d_in[6];
    const float* t2v_r_b = (const float*)d_in[7];
    const float* f_key_w = (const float*)d_in[8];
    const float* f_key_b = (const float*)d_in[9];
    const float* f_qry_w = (const float*)d_in[10];
    const float* f_qry_b = (const float*)d_in[11];
    const float* f_val_w = (const float*)d_in[12];
    const float* f_val_b = (const float*)d_in[13];
    const float* f_out_w = (const float*)d_in[14];
    const float* f_out_b = (const float*)d_in[15];
    const float* r_key_w = (const float*)d_in[16];
    const float* r_key_b = (const float*)d_in[17];
    const float* r_qry_w = (const float*)d_in[18];
    const float* r_qry_b = (const float*)d_in[19];
    const float* r_val_w = (const float*)d_in[20];
    const float* r_val_b = (const float*)d_in[21];
    const float* r_out_w = (const float*)d_in[22];
    const float* r_out_b = (const float*)d_in[23];
    const float* combo_w = (const float*)d_in[24];
    const float* combo_b = (const float*)d_in[25];
    const int*   len_f   = (const int*)d_in[26];
    const int*   len_r   = (const int*)d_in[27];

    float* f_ctx_out = (float*)d_ws;                       // [B, 256]
    float* r_ctx_out = f_ctx_out + (size_t)BATCH * ODIM;   // [B, 256]

    attn_kernel<<<BATCH, 256, 0, stream>>>(ts_f, feats_f, t2v_f_w, t2v_f_b,
                                           f_key_w, f_key_b, f_qry_w, f_qry_b,
                                           f_val_w, f_val_b, f_out_w, f_out_b,
                                           len_f, f_ctx_out);
    attn_kernel<<<BATCH, 256, 0, stream>>>(ts_r, feats_r, t2v_r_w, t2v_r_b,
                                           r_key_w, r_key_b, r_qry_w, r_qry_b,
                                           r_val_w, r_val_b, r_out_w, r_out_b,
                                           len_r, r_ctx_out);
    combo_kernel<<<BATCH, 128, 0, stream>>>(f_ctx_out, r_ctx_out, combo_w, combo_b,
                                            (float*)d_out);
}
